// Transformer_72043781423431
// MI455X (gfx1250) — matrix operand order repaired and checked by reading, not compiled, not run on hardware
//
#include <hip/hip_runtime.h>
#include <math.h>

// ---------------------------------------------------------------------------
// Problem constants
// ---------------------------------------------------------------------------
#define B_    64
#define N_    243
#define D_    512
#define H_    8
#define DH    64          // head dim
#define C3    1536        // 3*D
#define MLPH  1024
#define M_TOT (B_ * N_)   // 15552 rows = 64 * 243 (multiple of 64)
#define NP    256         // padded sequence length (per b,h tile)
#define BH    (B_ * H_)   // 512 attention batches

typedef __attribute__((ext_vector_type(16))) _Float16 v16h;
typedef __attribute__((ext_vector_type(8)))  float    v8f;

union Frag16 { v16h v; unsigned int u[8]; };

// ---------------------------------------------------------------------------
// Batched GEMM with B in transposed storage:  out = act(A[M,K] x B[K,N] + bias)
//   A : f16 row-major [M,K] (lda), per-batch stride aBatch (elements)
//   Bt: f16 row-major [N,K] (ldb = K stride) -- the logical B is Bt^T.
//   outF (f32) and/or outH (f16), leading dim ldc; optional bias/GELU/residual.
//
// Block: 128 threads (4 waves).  Block tile 64 x (32*NFRAG).
// Wave tile 32 x (16*NFRAG)  ->  2 x NFRAG WMMA accumulators per wave.
//   NFRAG=4: 64x128 block tile, 8 v_wmma per wave per K-step (dense GEMMs)
//   NFRAG=2: 64x64 block tile, 4 v_wmma (N=64 attention O-GEMM)
// K-step 32, double-buffered LDS: one barrier per K-step; next tile's global
// loads are held in registers across the WMMA burst.  Staging values are kept
// in NAMED uint4 scalars (not an array) so they stay in VGPRs -- the array
// form was spilled to scratch by the backend even at low pressure.
// ---------------------------------------------------------------------------
template <int NFRAG>
__global__ void __launch_bounds__(128)
__attribute__((amdgpu_waves_per_eu(1)))
gemm_bt_wmma_kernel(const _Float16* __restrict__ A, long aBatch, int lda,
                    const _Float16* __restrict__ Bt, long bBatch, int ldb,
                    const float* __restrict__ bias,
                    const float* __restrict__ resid, long rBatch,
                    float* __restrict__ outF, long oBatch,
                    _Float16* __restrict__ outH, long ohBatch,
                    int K, int ldc, int applyGelu)
{
    constexpr int BN = 32 * NFRAG;            // block tile N: 64 or 128
    __shared__ _Float16 As[2][64][40];        // [m][k], row padded to 40 halves
    __shared__ _Float16 Bs[2][BN][40];        // [n][k]

    const int tid  = threadIdx.x;
    const int lane = tid & 31;
    const int wave = tid >> 5;
    const int wm   = (wave >> 1) << 5;        // wave M offset (0/32)
    const int wn   = (wave & 1) * (16 * NFRAG);
    const int bm   = blockIdx.y << 6;
    const int bn   = blockIdx.x * BN;
    const int bz   = blockIdx.z;

    const _Float16* Ab = A  + (long)bz * aBatch;
    const _Float16* Bb = Bt + (long)bz * bBatch;

    v8f acc[2][NFRAG] = {};

    // staging assignments (16-byte vector loads everywhere)
    const int rowA = tid >> 1;                        // 0..63
    const int segA = (tid & 1) << 4;                  // 0 or 16 halves
    const int rowB = (NFRAG == 4) ? tid : (tid >> 1); // 0..BN-1
    const int segB = (NFRAG == 4) ? 0 : ((tid & 1) << 4);

    uint4 ra0, ra1, rb0, rb1, rb2, rb3;               // named scalars: no spill
    auto loadTile = [&](int k0) {
        const uint4* ap = (const uint4*)(Ab + (long)(bm + rowA) * lda + (k0 + segA));
        ra0 = ap[0]; ra1 = ap[1];
        const uint4* bp = (const uint4*)(Bb + (long)(bn + rowB) * ldb + (k0 + segB));
        rb0 = bp[0]; rb1 = bp[1];
        if constexpr (NFRAG == 4) { rb2 = bp[2]; rb3 = bp[3]; }
    };
    auto storeTile = [&](int buf) {
        *(uint4*)(&As[buf][rowA][segA])      = ra0;
        *(uint4*)(&As[buf][rowA][segA + 8])  = ra1;
        *(uint4*)(&Bs[buf][rowB][segB])      = rb0;
        *(uint4*)(&Bs[buf][rowB][segB + 8])  = rb1;
        if constexpr (NFRAG == 4) {
            *(uint4*)(&Bs[buf][rowB][16]) = rb2;
            *(uint4*)(&Bs[buf][rowB][24]) = rb3;
        }
    };

    const int mfrag = lane & 15;
    const int hh    = lane >> 4;
    const int nk    = K >> 5;

    loadTile(0);
    storeTile(0);

    for (int kt = 0; kt < nk; ++kt) {
        __syncthreads();                       // tile kt resident in buf kt&1
        if (kt + 1 < nk) loadTile((kt + 1) << 5);
        if (kt + 2 < nk)                       // L2 prefetch (global_prefetch_b8)
            __builtin_prefetch(Ab + (long)(bm + rowA) * lda + ((kt + 2) << 5) + segA, 0, 0);

        const int buf = kt & 1;
        // Fragment loads per CDNA5 16-bit WMMA layouts:
        //  A 16x32: lane m = lane%16; VGPR i<4: K=2i+8h ; i>=4: K=16+2(i-4)+8h
        //  B 32x16: lane n = lane%16; VGPR i<4: K=2i+8h ; i>=4: K=16+2(i-4)+8h
        Frag16 fa0, fa1;                       // hold both A fragments (16 VGPRs)
#pragma unroll
        for (int i = 0; i < 8; ++i) {
            const int ka = (i < 4) ? (2 * i + 8 * hh) : (16 + 2 * (i - 4) + 8 * hh);
            fa0.u[i] = *(const unsigned int*)(&As[buf][wm + mfrag][ka]);
            fa1.u[i] = *(const unsigned int*)(&As[buf][wm + 16 + mfrag][ka]);
        }
#pragma unroll
        for (int ni = 0; ni < NFRAG; ++ni) {   // stream B fragments (8 VGPRs)
            Frag16 fb;
#pragma unroll
            for (int i = 0; i < 8; ++i) {
                const int kb = (i < 4) ? (2 * i + 8 * hh) : (16 + 2 * (i - 4) + 8 * hh);
                fb.u[i] = *(const unsigned int*)(&Bs[buf][wn + 16 * ni + mfrag][kb]);
            }
            acc[0][ni] = __builtin_amdgcn_wmma_f32_16x16x32_f16(
                false, fa0.v, false, fb.v, (short)0, acc[0][ni], false, false);
            acc[1][ni] = __builtin_amdgcn_wmma_f32_16x16x32_f16(
                false, fa1.v, false, fb.v, (short)0, acc[1][ni], false, false);
        }

        if (kt + 1 < nk) storeTile((kt + 1) & 1);  // other buffer; safe: its
                                                   // readers synced at loop top
    }

    // epilogue: C/D layout: n = lane%16, m = vgpr_r + 8*(lane/16)
#pragma unroll
    for (int mi = 0; mi < 2; ++mi) {
#pragma unroll
        for (int ni = 0; ni < NFRAG; ++ni) {
#pragma unroll
            for (int r = 0; r < 8; ++r) {
                const int mm = bm + wm + mi * 16 + r + 8 * hh;
                const int nn = bn + wn + ni * 16 + mfrag;
                float v = acc[mi][ni][r];
                if (bias) v += bias[nn];
                if (applyGelu) v = 0.5f * v * (1.0f + erff(v * 0.70710678118654752f));
                const long idx = (long)mm * ldc + nn;
                if (resid) v += resid[(long)bz * rBatch + idx];
                if (outF)  outF[(long)bz * oBatch  + idx] = v;
                if (outH)  outH[(long)bz * ohBatch + idx] = (_Float16)v;
            }
        }
    }
}

// ---------------------------------------------------------------------------
// LayerNorm over last dim C (one block per row); optional f16 and f32 outputs
// ---------------------------------------------------------------------------
__global__ void __launch_bounds__(256)
ln_rows_kernel(const float* __restrict__ x, const float* __restrict__ g,
               const float* __restrict__ bta, int C,
               _Float16* __restrict__ out16, float* __restrict__ out32)
{
    const long row = blockIdx.x;
    const float* xr = x + row * C;
    __shared__ float red[256];
    const int t = threadIdx.x;

    float s = 0.f;
    for (int c = t; c < C; c += 256) s += xr[c];
    red[t] = s; __syncthreads();
    for (int o = 128; o > 0; o >>= 1) { if (t < o) red[t] += red[t + o]; __syncthreads(); }
    const float mean = red[0] / (float)C;
    __syncthreads();

    float s2 = 0.f;
    for (int c = t; c < C; c += 256) { float d = xr[c] - mean; s2 += d * d; }
    red[t] = s2; __syncthreads();
    for (int o = 128; o > 0; o >>= 1) { if (t < o) red[t] += red[t + o]; __syncthreads(); }
    const float rstd = rsqrtf(red[0] / (float)C + 1e-6f);

    for (int c = t; c < C; c += 256) {
        const float y = (xr[c] - mean) * rstd * g[c] + bta[c];
        if (out16) out16[row * C + c] = (_Float16)y;
        if (out32) out32[row * C + c] = y;
    }
}

// ---------------------------------------------------------------------------
// Softmax over 243 valid columns of a 256-wide row; writes zero-padded f16
// ---------------------------------------------------------------------------
__global__ void __launch_bounds__(256)
softmax_rows_kernel(const float* __restrict__ S, _Float16* __restrict__ P)
{
    const long base = (long)blockIdx.x * NP;
    const int t = threadIdx.x;
    __shared__ float red[256];

    const float v = (t < N_) ? S[base + t] : -3.0e38f;
    red[t] = v; __syncthreads();
    for (int o = 128; o > 0; o >>= 1) { if (t < o) red[t] = fmaxf(red[t], red[t + o]); __syncthreads(); }
    const float mx = red[0];
    __syncthreads();

    const float e = (t < N_) ? expf(v - mx) : 0.f;
    red[t] = e; __syncthreads();
    for (int o = 128; o > 0; o >>= 1) { if (t < o) red[t] += red[t + o]; __syncthreads(); }
    const float inv = 1.0f / red[0];

    P[base + t] = (_Float16)(e * inv);
}

// ---------------------------------------------------------------------------
// Elementwise helpers
// ---------------------------------------------------------------------------

// fp32 [K,N] row-major -> f16 [N,K] row-major (transposed weight conversion)
__global__ void cvt_t_f16_kernel(const float* __restrict__ s,
                                 _Float16* __restrict__ d, int K, int N)
{
    const long i = (long)blockIdx.x * blockDim.x + threadIdx.x;
    const long tot = (long)K * N;
    if (i >= tot) return;
    const int  n = (int)(i % N);
    const long k = i / N;
    d[(long)n * K + k] = (_Float16)s[i];
}

__global__ void addpos_kernel(const float* __restrict__ x,
                              const float* __restrict__ pos,
                              float* __restrict__ o)
{
    const long i = (long)blockIdx.x * blockDim.x + threadIdx.x;
    const long tot = (long)M_TOT * D_;
    if (i < tot) o[i] = x[i] + pos[i % ((long)N_ * D_)];
}

__global__ void concat3_kernel(const float* __restrict__ a, const float* __restrict__ b,
                               const float* __restrict__ c, float* __restrict__ o)
{
    const long i = (long)blockIdx.x * blockDim.x + threadIdx.x;
    const long tot = (long)M_TOT * C3;
    if (i >= tot) return;
    const int  cc = (int)(i % C3);
    const long m  = i / C3;
    float v;
    if (cc < D_)            v = a[m * D_ + cc];
    else if (cc < 2 * D_)   v = b[m * D_ + (cc - D_)];
    else                    v = c[m * D_ + (cc - 2 * D_)];
    o[i] = v;
}

__global__ void split3_kernel(const float* __restrict__ xc, float* __restrict__ a,
                              float* __restrict__ b, float* __restrict__ c)
{
    const long i = (long)blockIdx.x * blockDim.x + threadIdx.x;
    const long tot = (long)M_TOT * C3;
    if (i >= tot) return;
    const int  cc = (int)(i % C3);
    const long m  = i / C3;
    const float v = xc[i];
    if (cc < D_)            a[m * D_ + cc]            = v;
    else if (cc < 2 * D_)   b[m * D_ + (cc - D_)]     = v;
    else                    c[m * D_ + (cc - 2 * D_)] = v;
}

// src [B*N_, srcLd] f32 -> dst [B,H,NP,DH] f16 (rows >= N_ zero-padded)  (Q,K)
__global__ void pack_heads_kernel(const float* __restrict__ src, int srcLd,
                                  int colOff, float scale, _Float16* __restrict__ dst)
{
    const long i = (long)blockIdx.x * blockDim.x + threadIdx.x;
    const long tot = (long)BH * NP * DH;
    if (i >= tot) return;
    const int  dd = (int)(i & 63);
    const long r  = i >> 6;
    const int  nn = (int)(r & (NP - 1));
    const long r2 = r >> 8;              // b*H + h
    const int  h  = (int)(r2 & (H_ - 1));
    const int  b  = (int)(r2 >> 3);
    float v = 0.f;
    if (nn < N_)
        v = src[((long)(b * N_ + nn)) * srcLd + colOff + h * DH + dd] * scale;
    dst[i] = (_Float16)v;
}

// src [B*N_, srcLd] f32 -> dst [B,H,DH,NP] f16 (V stored transposed so the
// O = P*V GEMM can use vectorized [N,K] staging; cols >= N_ zero-padded)
__global__ void pack_v_t_kernel(const float* __restrict__ src, int srcLd,
                                int colOff, _Float16* __restrict__ dst)
{
    const long i = (long)blockIdx.x * blockDim.x + threadIdx.x;
    const long tot = (long)BH * DH * NP;
    if (i >= tot) return;
    const int  nn = (int)(i & (NP - 1));
    const long r  = i >> 8;
    const int  dd = (int)(r & 63);
    const long r2 = r >> 6;              // b*H + h
    const int  h  = (int)(r2 & (H_ - 1));
    const int  b  = (int)(r2 >> 3);
    float v = 0.f;
    if (nn < N_)
        v = src[((long)(b * N_ + nn)) * srcLd + colOff + h * DH + dd];
    dst[i] = (_Float16)v;
}

// O [BH,NP,DH] f32 -> dst [M_TOT, D_] f16 (heads interleaved back)
__global__ void unpack_o_kernel(const float* __restrict__ O, _Float16* __restrict__ dst)
{
    const long i = (long)blockIdx.x * blockDim.x + threadIdx.x;
    const long tot = (long)M_TOT * D_;
    if (i >= tot) return;
    const int  dd = (int)(i & 63);
    const long r  = i >> 6;
    const int  h  = (int)(r & 7);
    const long r2 = r >> 3;              // b*N_ + nn
    const int  nn = (int)(r2 % N_);
    const int  b  = (int)(r2 / N_);
    dst[i] = (_Float16)O[(((long)(b * H_ + h)) * NP + nn) * DH + dd];
}

// ---------------------------------------------------------------------------
// Host orchestration
// ---------------------------------------------------------------------------
extern "C" void kernel_launch(void* const* d_in, const int* in_sizes, int n_in,
                              void* d_out, int out_size, void* d_ws, size_t ws_size,
                              hipStream_t stream)
{
    (void)in_sizes; (void)n_in; (void)out_size; (void)ws_size;

    auto F = [&](int i) -> const float* { return (const float*)d_in[i]; };

    const float* x1in = F(0);
    const float* x2in = F(1);
    const float* x3in = F(2);

    // ---- workspace bump allocator (256B aligned) ----
    char*  wsc = (char*)d_ws;
    size_t off = 0;
    auto alloc = [&](size_t bytes) -> void* {
        off = (off + 255) & ~(size_t)255;
        void* p = wsc + off;
        off += bytes;
        return p;
    };

    float*    X1  = (float*)alloc((size_t)M_TOT * D_ * 4);
    float*    X2  = (float*)alloc((size_t)M_TOT * D_ * 4);
    float*    X3  = (float*)alloc((size_t)M_TOT * D_ * 4);
    float*    XC  = (float*)alloc((size_t)M_TOT * C3 * 4);
    float*    T0  = (float*)alloc((size_t)M_TOT * C3 * 4);
    _Float16* A16 = (_Float16*)alloc((size_t)M_TOT * C3 * 2);
    _Float16* B16 = (_Float16*)alloc((size_t)M_TOT * C3 * 2);
    _Float16* QP  = (_Float16*)alloc((size_t)BH * NP * DH * 2);
    _Float16* KP  = (_Float16*)alloc((size_t)BH * NP * DH * 2);
    _Float16* VPT = (_Float16*)alloc((size_t)BH * DH * NP * 2);
    float*    SC  = (float*)alloc((size_t)BH * NP * NP * 4);
    _Float16* PR  = (_Float16*)alloc((size_t)BH * NP * NP * 2);
    float*    OB  = (float*)alloc((size_t)BH * NP * DH * 4);

    auto EW = [&](long tot) -> dim3 { return dim3((unsigned)((tot + 255) / 256)); };

    // ---- convert fp32 weights [K,N] to transposed f16 [N,K] ----
    auto CVT = [&](int idx, int Kd, int Nd) -> _Float16* {
        _Float16* p = (_Float16*)alloc((size_t)Kd * Nd * 2);
        cvt_t_f16_kernel<<<EW((long)Kd * Nd), dim3(256), 0, stream>>>(F(idx), p, Kd, Nd);
        return p;
    };

    // Parameter leaf indices (jax pytree sorted-key order), base 3:
    //  chi: attn1{kb,kw,pb,pw,qb,qw,vb,vw}=3..10  attn2=11..18  attn3=19..26
    //       mlp{fc1_b,fc1_w,fc2_b,fc2_w}=27..30   n11_b..n33_g=31..48  nm_b=49 nm_g=50
    //  norm_b=51 norm_g=52  pos1..3=53..55
    //  shr[s] (base bs=56+24s): attnA{pb,pw,qkv_b,qkv_w}=bs+4a..  mlp=bs+12..15
    //       n1_b=bs+16 n1_g=bs+17 ... nm_b=bs+22 nm_g=bs+23
    _Float16 *qkvW[2][3], *pW[2][3], *f1W[2], *f2W[2];
    for (int s = 0; s < 2; ++s) {
        const int bs = 56 + 24 * s;
        for (int a = 0; a < 3; ++a) {
            pW[s][a]   = CVT(bs + 1 + 4 * a, D_, D_);
            qkvW[s][a] = CVT(bs + 3 + 4 * a, D_, C3);
        }
        f1W[s] = CVT(bs + 13, C3, MLPH);
        f2W[s] = CVT(bs + 15, MLPH, C3);
    }
    _Float16 *cKW[3], *cPW[3], *cQW[3], *cVW[3];
    for (int a = 0; a < 3; ++a) {
        const int i0 = 3 + 8 * a;
        cKW[a] = CVT(i0 + 1, D_, D_);
        cPW[a] = CVT(i0 + 3, D_, D_);
        cQW[a] = CVT(i0 + 5, D_, D_);
        cVW[a] = CVT(i0 + 7, D_, D_);
    }
    _Float16* cF1 = CVT(28, C3, MLPH);
    _Float16* cF2 = CVT(30, MLPH, C3);

    // ---- launch helpers ----
    auto GEMMB = [&](const _Float16* A, long aB, int lda,
                     const _Float16* Bt, long bB, int ldb,
                     const float* bias, const float* resid, long rB,
                     float* oF, long oB, _Float16* oH, long ohB,
                     int Mr, int Nc, int K, int ldc, int gelu, int batch) {
        dim3 grid((unsigned)(Nc / 128), (unsigned)(Mr / 64), (unsigned)batch);
        gemm_bt_wmma_kernel<4><<<grid, dim3(128), 0, stream>>>(
            A, aB, lda, Bt, bB, ldb, bias, resid, rB, oF, oB, oH, ohB, K, ldc, gelu);
    };
    auto GEMMS = [&](const _Float16* A, long aB, int lda,
                     const _Float16* Bt, long bB, int ldb,
                     const float* bias, const float* resid, long rB,
                     float* oF, long oB, _Float16* oH, long ohB,
                     int Mr, int Nc, int K, int ldc, int gelu, int batch) {
        dim3 grid((unsigned)(Nc / 64), (unsigned)(Mr / 64), (unsigned)batch);
        gemm_bt_wmma_kernel<2><<<grid, dim3(128), 0, stream>>>(
            A, aB, lda, Bt, bB, ldb, bias, resid, rB, oF, oB, oH, ohB, K, ldc, gelu);
    };
    auto LN = [&](const float* x, int gi, int bi, int C, _Float16* o16, float* o32) {
        ln_rows_kernel<<<dim3(M_TOT), dim3(256), 0, stream>>>(x, F(gi), F(bi), C, o16, o32);
    };

    const long tPK  = (long)BH * NP * DH;
    const long t512 = (long)M_TOT * D_;
    const long t1536 = (long)M_TOT * C3;

    // attention core: S = Q*K^T -> softmax -> O = P*V -> unpack to B16 [M,512] f16
    auto ATTN = [&]() {
        // S: A=QP[256,64], Bt=KP[256,64] (logical K^T[64,256])
        GEMMB(QP, (long)NP * DH, DH, KP, (long)NP * DH, DH, nullptr, nullptr, 0,
              SC, (long)NP * NP, nullptr, 0, NP, NP, DH, NP, 0, BH);
        softmax_rows_kernel<<<dim3(BH * NP), dim3(256), 0, stream>>>(SC, PR);
        // O: A=PR[256,256], Bt=VPT[64,256] (logical V[256,64]); N=64 -> small tile
        GEMMS(PR, (long)NP * NP, NP, VPT, (long)DH * NP, NP, nullptr, nullptr, 0,
              OB, (long)NP * DH, nullptr, 0, NP, DH, NP, DH, 0, BH);
        unpack_o_kernel<<<EW(t512), dim3(256), 0, stream>>>(OB, B16);
    };

    // self attention with fused residual into x
    auto SELF = [&](float* x, int gI, int bI, int pbI, _Float16* pw16,
                    int qkvbI, _Float16* qkvw16) {
        LN(x, gI, bI, D_, A16, nullptr);
        GEMMB(A16, 0, D_, qkvw16, 0, D_, F(qkvbI), nullptr, 0,
              T0, 0, nullptr, 0, M_TOT, C3, D_, C3, 0, 1);
        pack_heads_kernel<<<EW(tPK), dim3(256), 0, stream>>>(T0, C3, 0,  0.125f, QP);
        pack_heads_kernel<<<EW(tPK), dim3(256), 0, stream>>>(T0, C3, D_, 1.0f,   KP);
        pack_v_t_kernel  <<<EW(tPK), dim3(256), 0, stream>>>(T0, C3, 2 * D_,     VPT);
        ATTN();
        GEMMB(B16, 0, D_, pw16, 0, D_, F(pbI), x, 0,
              x, 0, nullptr, 0, M_TOT, D_, D_, D_, 0, 1);
    };

    // cross attention: q from qx, k from kx, v from vx, residual into tgt
    auto CROSS = [&](const float* qx, int qg, int qb2, const float* kx, int kg, int kb2,
                     const float* vx, int vg, int vb2, float* tgt,
                     int qbI, _Float16* qw, int kbI, _Float16* kw,
                     int vbI, _Float16* vw, int pbI, _Float16* pw) {
        LN(qx, qg, qb2, D_, A16, nullptr);
        GEMMB(A16, 0, D_, qw, 0, D_, F(qbI), nullptr, 0, T0, 0, nullptr, 0,
              M_TOT, D_, D_, D_, 0, 1);
        pack_heads_kernel<<<EW(tPK), dim3(256), 0, stream>>>(T0, D_, 0, 0.125f, QP);
        LN(kx, kg, kb2, D_, A16, nullptr);
        GEMMB(A16, 0, D_, kw, 0, D_, F(kbI), nullptr, 0, T0, 0, nullptr, 0,
              M_TOT, D_, D_, D_, 0, 1);
        pack_heads_kernel<<<EW(tPK), dim3(256), 0, stream>>>(T0, D_, 0, 1.0f, KP);
        LN(vx, vg, vb2, D_, A16, nullptr);
        GEMMB(A16, 0, D_, vw, 0, D_, F(vbI), nullptr, 0, T0, 0, nullptr, 0,
              M_TOT, D_, D_, D_, 0, 1);
        pack_v_t_kernel<<<EW(tPK), dim3(256), 0, stream>>>(T0, D_, 0, VPT);
        ATTN();
        GEMMB(B16, 0, D_, pw, 0, D_, F(pbI), tgt, 0,
              tgt, 0, nullptr, 0, M_TOT, D_, D_, D_, 0, 1);
    };

    // MLP on concat(x1,x2,x3), fused exact-erf GELU, residual; then split back
    auto MLP = [&](int f1bI, _Float16* f1w, int f2bI, _Float16* f2w, int nmgI, int nmbI) {
        concat3_kernel<<<EW(t1536), dim3(256), 0, stream>>>(X1, X2, X3, XC);
        LN(XC, nmgI, nmbI, C3, A16, nullptr);
        GEMMB(A16, 0, C3, f1w, 0, C3, F(f1bI), nullptr, 0,
              nullptr, 0, B16, 0, M_TOT, MLPH, C3, MLPH, 1, 1);
        GEMMB(B16, 0, MLPH, f2w, 0, MLPH, F(f2bI), XC, 0,
              XC, 0, nullptr, 0, M_TOT, C3, MLPH, C3, 0, 1);
        split3_kernel<<<EW(t1536), dim3(256), 0, stream>>>(XC, X1, X2, X3);
    };

    // ---- forward pass ----
    addpos_kernel<<<EW(t512), dim3(256), 0, stream>>>(x1in, F(53), X1);
    addpos_kernel<<<EW(t512), dim3(256), 0, stream>>>(x2in, F(54), X2);
    addpos_kernel<<<EW(t512), dim3(256), 0, stream>>>(x3in, F(55), X3);

    for (int s = 0; s < 2; ++s) {
        const int bs = 56 + 24 * s;
        SELF(X1, bs + 17, bs + 16, bs + 0,  pW[s][0], bs + 2,  qkvW[s][0]);
        SELF(X2, bs + 19, bs + 18, bs + 4,  pW[s][1], bs + 6,  qkvW[s][1]);
        SELF(X3, bs + 21, bs + 20, bs + 8,  pW[s][2], bs + 10, qkvW[s][2]);
        MLP(bs + 12, f1W[s], bs + 14, f2W[s], bs + 23, bs + 22);
    }

    // chi block (sequential, uses updated streams like the reference)
    CROSS(X2, 32, 31, X3, 34, 33, X1, 36, 35, X1,
          7,  cQW[0], 3,  cKW[0], 9,  cVW[0], 5,  cPW[0]);
    CROSS(X1, 38, 37, X3, 40, 39, X2, 42, 41, X2,
          15, cQW[1], 11, cKW[1], 17, cVW[1], 13, cPW[1]);
    CROSS(X1, 44, 43, X2, 46, 45, X3, 48, 47, X3,
          23, cQW[2], 19, cKW[2], 25, cVW[2], 21, cPW[2]);
    MLP(27, cF1, 29, cF2, 50, 49);

    // final concat + LayerNorm -> fp32 output
    concat3_kernel<<<EW(t1536), dim3(256), 0, stream>>>(X1, X2, X3, XC);
    ln_rows_kernel<<<dim3(M_TOT), dim3(256), 0, stream>>>(
        XC, F(52), F(51), C3, nullptr, (float*)d_out);
}
